// TopKformer_45853070852235
// MI455X (gfx1250) — matrix operand order repaired and checked by reading, not compiled, not run on hardware
//
#include <hip/hip_runtime.h>

#define DD 64

typedef __attribute__((ext_vector_type(16))) _Float16 v16h;
typedef __attribute__((ext_vector_type(8)))  float    v8f;

// ---------------------------------------------------------------------------
// Kernel 1: z = x / max(||x||,eps), stored as f16 (halves gather bytes; the
// 2M edge-gathers read z, and f16 is the WMMA operand type anyway).
// One wave (32 lanes) per row, float2 per lane, shuffle reduction.
// ---------------------------------------------------------------------------
__global__ void __launch_bounds__(256) normalize_f16_kernel(
    const float* __restrict__ x, _Float16* __restrict__ z16, int n_rows) {
  const int lane = threadIdx.x & 31;
  const int row  = (int)(((size_t)blockIdx.x * blockDim.x + threadIdx.x) >> 5);
  if (row >= n_rows) return;

  const float2 v = *(const float2*)(x + (size_t)row * DD + lane * 2);
  float ss = v.x * v.x + v.y * v.y;
#pragma unroll
  for (int off = 16; off > 0; off >>= 1) ss += __shfl_xor(ss, off, 32);
  const float inv = 1.0f / fmaxf(sqrtf(ss), 1e-12f);

  union { _Float16 h[2]; unsigned int u; } p;
  p.h[0] = (_Float16)(v.x * inv);
  p.h[1] = (_Float16)(v.y * inv);
  *(unsigned int*)(z16 + (size_t)row * DD + lane * 2) = p.u;
}

// ---------------------------------------------------------------------------
// Kernel 2: one wave handles 16 edges.
//  - gather 16 zu rows (A, 16x64 f16) and 16 zi rows (B, 64x16 f16)
//  - two v_wmma_f32_16x16x32_f16 accumulate the 16x16 pairwise-dot tile
//  - extract diagonal -> per-edge score s, compute w on lanes 0..15
//  - all 32 lanes scatter w*xi[i] -> out[u], w*xu[u] -> out[N_USERS+i]
//    via global_atomic_add_f32 (L2-resident).
// ---------------------------------------------------------------------------
__global__ void __launch_bounds__(256) edge_wmma_kernel(
    const float* __restrict__ x,   const float* __restrict__ beta,
    const float* __restrict__ du,  const float* __restrict__ di,
    const int*   __restrict__ u,   const int*   __restrict__ it,
    const _Float16* __restrict__ z16,
    float* __restrict__ out, int n_users, int n_edges) {
  const int lane = threadIdx.x & 31;
  const int wave = (int)(((size_t)blockIdx.x * blockDim.x + threadIdx.x) >> 5);
  const long base = (long)wave * 16;
  if (base >= n_edges) return;           // wave-uniform: EXEC stays all-ones

  int eu = 0, ei = 0;
  const bool edge_ok = (lane < 16) && (base + lane < n_edges);
  if (edge_ok) { eu = u[base + lane]; ei = it[base + lane]; }

  // Row indices for the matrix gather: lane l and l+16 both load edge (l&15).
  const int myU = __shfl(eu, lane & 15, 32);
  const int myI = __shfl(ei, lane & 15, 32);
  const int hi  = lane >> 4;             // 0: lanes 0-15, 1: lanes 16-31

  const _Float16* za = z16 + (size_t)myU * DD;               // user rows (A)
  const _Float16* zb = z16 + (size_t)(n_users + myI) * DD;   // item rows (B)

  union Op { v16h v; float4 f4[2]; } a0, a1, b0, b1;
  // A (16-bit 16x32): low lanes K{0..7,16..23}; high lanes K{8..15,24..31}
  a0.f4[0] = *(const float4*)(za +      hi * 8);
  a0.f4[1] = *(const float4*)(za + 16 + hi * 8);
  a1.f4[0] = *(const float4*)(za + 32 + hi * 8);
  a1.f4[1] = *(const float4*)(za + 48 + hi * 8);
  // B (16-bit 32x16): low lanes K{0..7,16..23}; high lanes K{8..15,24..31}
  b0.f4[0] = *(const float4*)(zb +      hi * 8);
  b0.f4[1] = *(const float4*)(zb + 16 + hi * 8);
  b1.f4[0] = *(const float4*)(zb + 32 + hi * 8);
  b1.f4[1] = *(const float4*)(zb + 48 + hi * 8);

  v8f c = {};
  c = __builtin_amdgcn_wmma_f32_16x16x32_f16(false, a0.v, false, b0.v,
                                             (short)0, c, false, false);
  c = __builtin_amdgcn_wmma_f32_16x16x32_f16(false, a1.v, false, b1.v,
                                             (short)0, c, false, false);

  // Diagonal C[d][d]: d<8 at (vgpr d, lane d); d>=8 at (vgpr d-8, lane d+16).
  // So on lanes 0-7 and 24-31 the wanted component index is (lane & 7).
  const int k = lane & 7;
  float dv = c[0];
  dv = (k == 1) ? c[1] : dv;
  dv = (k == 2) ? c[2] : dv;
  dv = (k == 3) ? c[3] : dv;
  dv = (k == 4) ? c[4] : dv;
  dv = (k == 5) ? c[5] : dv;
  dv = (k == 6) ? c[6] : dv;
  dv = (k == 7) ? c[7] : dv;
  // Gather: edge e lives on source lane e (e<8) or e+16 (e>=8).
  float s = __shfl(dv, (lane < 8) ? lane : (lane + 16), 32);

  float w = 0.0f;
  if (edge_ok) {
    s -= beta[eu];
    const float sg = 1.0f / (1.0f + __expf(-s));
    w = 4.0f * sg * (1.0f - sg);
    w = w / sqrtf(du[eu]) / sqrtf(di[ei]);
  }

  const int col = lane * 2;
#pragma unroll 4
  for (int e = 0; e < 16; ++e) {
    const float we = __shfl(w,  e, 32);
    const int   uu = __shfl(eu, e, 32);
    const int   ix = __shfl(ei, e, 32);
    const float2 xiv = *(const float2*)(x + (size_t)(n_users + ix) * DD + col);
    const float2 xuv = *(const float2*)(x + (size_t)uu * DD + col);
    float* ou = out + (size_t)uu * DD + col;               // dxu[u] += w*xi[i]
    float* oi = out + (size_t)(n_users + ix) * DD + col;   // dxi[i] += w*xu[u]
    atomicAdd(ou,     we * xiv.x);
    atomicAdd(ou + 1, we * xiv.y);
    atomicAdd(oi,     we * xuv.x);
    atomicAdd(oi + 1, we * xuv.y);
  }
}

extern "C" void kernel_launch(void* const* d_in, const int* in_sizes, int n_in,
                              void* d_out, int out_size, void* d_ws, size_t ws_size,
                              hipStream_t stream) {
  const float* x    = (const float*)d_in[0];
  const float* beta = (const float*)d_in[1];
  const float* du   = (const float*)d_in[2];
  const float* di   = (const float*)d_in[3];
  const int*   u    = (const int*)d_in[4];
  const int*   it   = (const int*)d_in[5];

  const int n_rows  = in_sizes[0] / DD;   // 150000
  const int n_users = in_sizes[1];        // 100000
  const int n_edges = in_sizes[4];        // 2000000

  float*     out = (float*)d_out;
  _Float16*  z16 = (_Float16*)d_ws;       // 150000*64*2 = 19.2 MB scratch

  // out is poisoned by the harness; atomics need zeros.
  hipMemsetAsync(d_out, 0, (size_t)out_size * sizeof(float), stream);

  {
    const int block = 256;                               // 8 waves/block
    const int grid  = (n_rows * 32 + block - 1) / block; // 1 wave per row
    normalize_f16_kernel<<<grid, block, 0, stream>>>(x, z16, n_rows);
  }
  {
    const int groups = (n_edges + 15) / 16;              // 16 edges per wave
    const int block  = 256;
    const int grid   = (int)(((size_t)groups * 32 + block - 1) / block);
    edge_wmma_kernel<<<grid, block, 0, stream>>>(x, beta, du, di, u, it,
                                                 z16, out, n_users, n_edges);
  }
}
